// HierarchyStructure_42958262895201
// MI455X (gfx1250) — compile-verified, matrix operand order checked by reading, not run on hardware
//
#include <hip/hip_runtime.h>

typedef __attribute__((ext_vector_type(16))) _Float16 v16h;
typedef __attribute__((ext_vector_type(8)))  _Float16 v8h;
typedef __attribute__((ext_vector_type(8)))  float    v8f;

#define Bc 8
#define Np 4096
#define NPOINT 1024
#define NSAMPLE 32
#define DPTS 61
#define CIN 64
#define M0 128
#define M1 256
#define M2 512

// ---------------------------------------------------------------- utilities
__global__ void zero_kernel(float* p, int n) {
  int i = blockIdx.x * blockDim.x + threadIdx.x;
  if (i < n) p[i] = 0.f;
}

// new_xyz gather -> d_out part 1, [B, NPOINT, 3]
__global__ void newxyz_kernel(const float* __restrict__ xyz,
                              const int* __restrict__ cent,
                              float* __restrict__ out) {
  int i = blockIdx.x * blockDim.x + threadIdx.x;
  if (i >= Bc * NPOINT) return;
  int b = i >> 10;
  int c = cent[i];
  const float* xb = xyz + (size_t)b * Np * 3;
  out[(size_t)i * 3 + 0] = xb[c * 3 + 0];
  out[(size_t)i * 3 + 1] = xb[c * 3 + 1];
  out[(size_t)i * 3 + 2] = xb[c * 3 + 2];
}

// ---------------------------------------------------------------- KNN top-32
__global__ __launch_bounds__(64)
void knn_kernel(const float* __restrict__ xyz, const int* __restrict__ cent,
                int* __restrict__ idx) {
  __shared__ float dist[Np];
  __shared__ float rv[64];
  __shared__ int   ri[64];
  int b = blockIdx.x >> 10, p = blockIdx.x & 1023, t = threadIdx.x;
  int c = cent[b * NPOINT + p];
  const float* xb = xyz + (size_t)b * Np * 3;
  float cx = xb[c * 3], cy = xb[c * 3 + 1], cz = xb[c * 3 + 2];
  for (int n = t; n < Np; n += 64) {
    float dx = xb[n * 3] - cx, dy = xb[n * 3 + 1] - cy, dz = xb[n * 3 + 2] - cz;
    dist[n] = dx * dx + dy * dy + dz * dz;
  }
  __syncthreads();
  int* out = idx + (size_t)(b * NPOINT + p) * NSAMPLE;
  for (int k = 0; k < NSAMPLE; k++) {
    float bv = 3.4e38f; int bi = 0;
    for (int n = t; n < Np; n += 64) {
      float d = dist[n];
      if (d < bv) { bv = d; bi = n; }
    }
    rv[t] = bv; ri[t] = bi;
    __syncthreads();
    if (t == 0) {
      float m = rv[0]; int mi = ri[0];
      for (int j = 1; j < 64; j++) if (rv[j] < m) { m = rv[j]; mi = ri[j]; }
      out[k] = mi;
      dist[mi] = 3.4e38f;
    }
    __syncthreads();
  }
}

// ------------------------------------------- gather + standardize -> fp16 X
__global__ __launch_bounds__(64)
void group_std_kernel(const float* __restrict__ xyz,
                      const float* __restrict__ points,
                      const int* __restrict__ cent,
                      const int* __restrict__ idx,
                      _Float16* __restrict__ X) {
  __shared__ float vals[CIN * NSAMPLE];
  __shared__ float rs[64], rq[64];
  __shared__ int sidx[NSAMPLE];
  int b = blockIdx.x >> 10, p = blockIdx.x & 1023, t = threadIdx.x;
  const int* ip = idx + (size_t)(b * NPOINT + p) * NSAMPLE;
  if (t < NSAMPLE) sidx[t] = ip[t];
  int c = cent[b * NPOINT + p];
  const float* xb = xyz + (size_t)b * Np * 3;
  float c0 = xb[c * 3], c1 = xb[c * 3 + 1], c2 = xb[c * 3 + 2];
  __syncthreads();
  float s = 0.f, q = 0.f;
  for (int sn = 0; sn < NSAMPLE; sn++) {
    int n = sidx[sn];
    float v;
    if (t < 3) {
      float cc = (t == 0) ? c0 : (t == 1 ? c1 : c2);
      v = xb[n * 3 + t] - cc;
    } else {
      v = points[((size_t)b * DPTS + (t - 3)) * Np + n];
    }
    vals[t * NSAMPLE + sn] = v;
    s += v; q += v * v;
  }
  rs[t] = s; rq[t] = q;
  __syncthreads();
  for (int st = 32; st > 0; st >>= 1) {
    if (t < st) { rs[t] += rs[t + st]; rq[t] += rq[t + st]; }
    __syncthreads();
  }
  float mean = rs[0] * (1.f / (CIN * NSAMPLE));
  float var = rq[0] * (1.f / (CIN * NSAMPLE)) - mean * mean;
  float sd = fmaxf(sqrtf(fmaxf(var, 0.f)), 1e-6f);
  float inv = 1.f / sd;
  _Float16* xr = X + ((size_t)(b * NPOINT + p) * NSAMPLE) * CIN;
  for (int sn = 0; sn < NSAMPLE; sn++)
    xr[sn * CIN + t] = (_Float16)((vals[t * NSAMPLE + sn] - mean) * inv);
}

// ---------------------------------------- per-channel sums for BN (training)
// X: [L, C] row-major fp16, C in {64,128,256} (divides 256)
__global__ __launch_bounds__(256)
void channel_stats(const _Float16* __restrict__ X, int L, int C,
                   float* __restrict__ sum, float* __restrict__ sumsq) {
  __shared__ float ls[256], lq[256];
  int t = threadIdx.x;
  if (t < C) { ls[t] = 0.f; lq[t] = 0.f; }
  __syncthreads();
  const long ROWS = 128;
  long base = (long)blockIdx.x * ROWS;
  long lim = ((long)L - base) * C;
  long nelem = ROWS * C;
  if (lim > nelem) lim = nelem;
  const _Float16* p = X + base * C;
  float s = 0.f, q = 0.f;
  for (long e = t; e < lim; e += 256) {
    float v = (float)p[e];
    s += v; q += v * v;
  }
  atomicAdd(&ls[t % C], s);
  atomicAdd(&lq[t % C], q);
  __syncthreads();
  if (t < C) { atomicAdd(&sum[t], ls[t]); atomicAdd(&sumsq[t], lq[t]); }
}

// ------------------- fold BN into weights: W'[o,i]=W*s_i, b'[o]=b+Σ W*t_i
__global__ __launch_bounds__(256)
void fold_kernel(const float* __restrict__ W, const float* __restrict__ b,
                 const float* __restrict__ gamma, const float* __restrict__ beta,
                 const float* __restrict__ sum, const float* __restrict__ sumsq,
                 float cnt, int Ci, _Float16* __restrict__ Wf,
                 float* __restrict__ bf, int has_bn) {
  __shared__ float red[256];
  int o = blockIdx.x, t = threadIdx.x;
  float part = 0.f;
  for (int i = t; i < Ci; i += 256) {
    float sc = 1.f, sh = 0.f;
    if (has_bn) {
      float m = sum[i] / cnt;
      float v = sumsq[i] / cnt - m * m;
      sc = gamma[i] * rsqrtf(v + 1e-5f);
      sh = beta[i] - m * sc;
    }
    float w = W[(size_t)o * Ci + i];
    Wf[(size_t)o * Ci + i] = (_Float16)(w * sc);
    part += w * sh;
  }
  red[t] = part;
  __syncthreads();
  for (int st = 128; st > 0; st >>= 1) {
    if (t < st) red[t] += red[t + st];
    __syncthreads();
  }
  if (t == 0) bf[o] = b[o] + red[0];
}

// --------------------------------------------------------- WMMA GEMM engine
// C = relu(A[M,K] * X[L,K]^T + bias (+ A2[M,K2] * X2[L,K2]^T + bias2))
// A,X row-major fp16.  One WAVE owns one 16-column tile and iterates over ALL
// row tiles, holding every B fragment (K/32 x v16h) in registers so each X
// element is fetched from HBM exactly once; A fragments re-read per row tile
// but the folded weights (<=256KB) live in the 192MB L2.
// OUTH: store fp16 [L, M] row-major; else fp32 [b, M, P] (col n -> b=n/P,p=n%P).
__device__ __forceinline__ v16h load_frag(const _Float16* row, int k0, int g) {
  v8h a0 = *(const v8h*)(row + k0 + g * 8);
  v8h a1 = *(const v8h*)(row + k0 + 16 + g * 8);
  return __builtin_shufflevector(a0, a1, 0, 1, 2, 3, 4, 5, 6, 7,
                                 8, 9, 10, 11, 12, 13, 14, 15);
}

template <int K, int K2, bool OUTH>
__global__ __launch_bounds__(256)
void gemm_wmma_kernel(const _Float16* __restrict__ A,
                      const _Float16* __restrict__ X,
                      const _Float16* __restrict__ A2,
                      const _Float16* __restrict__ X2,
                      const float* __restrict__ bias,
                      const float* __restrict__ bias2,
                      _Float16* __restrict__ outH,
                      float* __restrict__ outF,
                      int M, int L, int P) {
  int ct = (blockIdx.x * 256 + threadIdx.x) >> 5;   // column tile index
  if (ct >= (L >> 4)) return;
  int lane = threadIdx.x & 31, lm = lane & 15, g = lane >> 4;

  // Load the whole B-operand panel for this column tile into registers.
  const _Float16* xrow = X + (size_t)(ct * 16 + lm) * K;
  v16h bfrag[K / 32];
#pragma unroll
  for (int k = 0; k < K / 32; k++) bfrag[k] = load_frag(xrow, k * 32, g);

  v16h bfrag2[(K2 > 0 ? K2 / 32 : 1)];
  if constexpr (K2 > 0) {
    const _Float16* x2row = X2 + (size_t)(ct * 16 + lm) * K2;
#pragma unroll
    for (int k = 0; k < K2 / 32; k++) bfrag2[k] = load_frag(x2row, k * 32, g);
  }

  int n = ct * 16 + lm;          // output column
  int nrt = M >> 4;
  for (int rt = 0; rt < nrt; rt++) {
    if (rt + 1 < nrt)  // prefetch next row-tile's weights (global_prefetch_b8)
      __builtin_prefetch(A + (size_t)((rt + 1) * 16 + lm) * K, 0, 0);
    v8f c = {};
    const _Float16* arow = A + (size_t)(rt * 16 + lm) * K;
#pragma unroll
    for (int k = 0; k < K / 32; k++) {
      v16h a = load_frag(arow, k * 32, g);
      c = __builtin_amdgcn_wmma_f32_16x16x32_f16(false, a, false, bfrag[k],
                                                 (short)0, c, false, false);
    }
    if constexpr (K2 > 0) {
      const _Float16* a2row = A2 + (size_t)(rt * 16 + lm) * K2;
#pragma unroll
      for (int k = 0; k < K2 / 32; k++) {
        v16h a = load_frag(a2row, k * 32, g);
        c = __builtin_amdgcn_wmma_f32_16x16x32_f16(false, a, false, bfrag2[k],
                                                   (short)0, c, false, false);
      }
    }
    int m0 = rt * 16 + g * 8;    // first of 8 consecutive rows in this lane
    float vals[8];
#pragma unroll
    for (int r = 0; r < 8; r++) {
      float v = c[r] + bias[m0 + r];
      if constexpr (K2 > 0) v += bias2[m0 + r];
      vals[r] = fmaxf(v, 0.f);
    }
    if constexpr (OUTH) {
      v8h h;
#pragma unroll
      for (int r = 0; r < 8; r++) h[r] = (_Float16)vals[r];
      *(v8h*)(outH + (size_t)n * M + m0) = h;
    } else {
      int bcol = n / P, pcol = n % P;
      float* base = outF + ((size_t)bcol * M) * P + pcol;
#pragma unroll
      for (int r = 0; r < 8; r++) base[(size_t)(m0 + r) * P] = vals[r];
    }
  }
}

// -------------------------------- attention: logits -> softmax -> aggregate
__global__ __launch_bounds__(256)
void attention_kernel(const _Float16* __restrict__ feat,  // [L, 256]
                      const _Float16* __restrict__ H,     // [L, 64]
                      const float* __restrict__ Wa2,      // [64]
                      const float* __restrict__ ba2,      // [1]
                      _Float16* __restrict__ agg) {       // [G, 256]
  __shared__ float w[NSAMPLE];
  int gi = blockIdx.x, t = threadIdx.x;
  size_t colbase = (size_t)gi * NSAMPLE;
  if (t < NSAMPLE) {
    float acc = ba2[0];
    const _Float16* hcol = H + (colbase + t) * 64;
    for (int i = 0; i < 64; i++) acc += Wa2[i] * (float)hcol[i];
    w[t] = acc;
  }
  __syncthreads();
  if (t < NSAMPLE) {
    float v = w[t];
    float mx = v;
    for (int o = 16; o > 0; o >>= 1) mx = fmaxf(mx, __shfl_xor(mx, o, 32));
    float e = __expf(v - mx);
    float sm = e;
    for (int o = 16; o > 0; o >>= 1) sm += __shfl_xor(sm, o, 32);
    w[t] = e / sm;
  }
  __syncthreads();
  float acc = 0.f;
  for (int s = 0; s < NSAMPLE; s++)
    acc += (float)feat[(colbase + s) * M1 + t] * w[s];
  agg[(size_t)gi * M1 + t] = (_Float16)acc;
}

// ---------------------------------------------------------------- launcher
extern "C" void kernel_launch(void* const* d_in, const int* in_sizes, int n_in,
                              void* d_out, int out_size, void* d_ws, size_t ws_size,
                              hipStream_t stream) {
  const float* xyz    = (const float*)d_in[0];
  const float* points = (const float*)d_in[1];
  const int*   cent   = (const int*)  d_in[2];
  const float* gamma0 = (const float*)d_in[3];
  const float* beta0  = (const float*)d_in[4];
  const float* W1     = (const float*)d_in[5];
  const float* b1     = (const float*)d_in[6];
  const float* gamma1 = (const float*)d_in[7];
  const float* beta1  = (const float*)d_in[8];
  const float* W2     = (const float*)d_in[9];
  const float* b2     = (const float*)d_in[10];
  const float* Wres   = (const float*)d_in[11];
  const float* bres   = (const float*)d_in[12];
  const float* Wa1    = (const float*)d_in[13];
  const float* ba1    = (const float*)d_in[14];
  const float* Wa2    = (const float*)d_in[15];
  const float* ba2    = (const float*)d_in[16];
  const float* gamma2 = (const float*)d_in[17];
  const float* beta2  = (const float*)d_in[18];
  const float* Wg     = (const float*)d_in[19];
  const float* bg     = (const float*)d_in[20];

  const long Lfull = (long)Bc * NPOINT * NSAMPLE;  // 262144
  const long Lagg  = (long)Bc * NPOINT;            // 8192

  char* ws = (char*)d_ws;
  size_t off = 0;
  auto carve = [&](size_t bytes) -> char* {
    char* r = ws + off;
    off = (off + bytes + 255) & ~(size_t)255;
    return r;
  };
  int*      idxbuf = (int*)     carve(Lagg * NSAMPLE * 4);
  _Float16* Xb     = (_Float16*)carve(Lfull * CIN * 2);
  _Float16* interb = (_Float16*)carve(Lfull * M0 * 2);
  _Float16* featb  = (_Float16*)carve(Lfull * M1 * 2);
  _Float16* Hb     = (_Float16*)carve(Lfull * 64 * 2);
  _Float16* aggb   = (_Float16*)carve(Lagg * M1 * 2);
  float*    stats  = (float*)   carve(896 * 4);
  float *s0 = stats, *q0 = stats + 64, *s1 = stats + 128, *q1 = stats + 256,
        *s2 = stats + 384, *q2 = stats + 640;
  _Float16* W1f   = (_Float16*)carve(M0 * CIN * 2); float* b1f   = (float*)carve(M0 * 4);
  _Float16* W2f   = (_Float16*)carve(M1 * M0 * 2);  float* b2f   = (float*)carve(M1 * 4);
  _Float16* Wresf = (_Float16*)carve(M1 * CIN * 2); float* bresf = (float*)carve(M1 * 4);
  _Float16* Wa1f  = (_Float16*)carve(64 * M1 * 2);  float* ba1f  = (float*)carve(64 * 4);
  _Float16* Wgf   = (_Float16*)carve(M2 * M1 * 2);  float* bgf   = (float*)carve(M2 * 4);

  float* outNew = (float*)d_out;                            // [B, NPOINT, 3]
  float* outGf  = (float*)d_out + (size_t)Bc * NPOINT * 3;  // [B, M2, NPOINT]

  zero_kernel<<<4, 256, 0, stream>>>(stats, 896);
  newxyz_kernel<<<32, 256, 0, stream>>>(xyz, cent, outNew);
  knn_kernel<<<Bc * NPOINT, 64, 0, stream>>>(xyz, cent, idxbuf);
  group_std_kernel<<<Bc * NPOINT, 64, 0, stream>>>(xyz, points, cent, idxbuf, Xb);

  // BN0 stats over standardized input, fold into W1, GEMM1 (relu)
  channel_stats<<<(int)(Lfull / 128), 256, 0, stream>>>(Xb, (int)Lfull, CIN, s0, q0);
  fold_kernel<<<M0, 256, 0, stream>>>(W1, b1, gamma0, beta0, s0, q0,
                                      (float)Lfull, CIN, W1f, b1f, 1);
  gemm_wmma_kernel<CIN, 0, true><<<(int)(Lfull / 16 / 8), 256, 0, stream>>>(
      W1f, Xb, nullptr, nullptr, b1f, nullptr, interb, nullptr,
      M0, (int)Lfull, 0);

  // BN1 stats over inter, fold W2; convert residual + attention weights
  channel_stats<<<(int)(Lfull / 128), 256, 0, stream>>>(interb, (int)Lfull, M0, s1, q1);
  fold_kernel<<<M1, 256, 0, stream>>>(W2, b2, gamma1, beta1, s1, q1,
                                      (float)Lfull, M0, W2f, b2f, 1);
  fold_kernel<<<M1, 256, 0, stream>>>(Wres, bres, gamma1, beta1, s1, q1,
                                      (float)Lfull, CIN, Wresf, bresf, 0);
  fold_kernel<<<64, 256, 0, stream>>>(Wa1, ba1, gamma1, beta1, s1, q1,
                                      (float)Lfull, M1, Wa1f, ba1f, 0);

  // feat = relu(W2'*inter + b2' + Wres*x + bres)   (dual-K fused GEMM)
  gemm_wmma_kernel<M0, CIN, true><<<(int)(Lfull / 16 / 8), 256, 0, stream>>>(
      W2f, interb, Wresf, Xb, b2f, bresf, featb, nullptr,
      M1, (int)Lfull, 0);

  // H = relu(Wa1*feat + ba1)
  gemm_wmma_kernel<M1, 0, true><<<(int)(Lfull / 16 / 8), 256, 0, stream>>>(
      Wa1f, featb, nullptr, nullptr, ba1f, nullptr, Hb, nullptr,
      64, (int)Lfull, 0);

  // attention softmax + aggregation -> agg [G, 256] fp16
  attention_kernel<<<Bc * NPOINT, 256, 0, stream>>>(featb, Hb, Wa2, ba2, aggb);

  // BN2 stats over agg, fold into Wg, final GEMM -> f32 output [B, 512, NPOINT]
  channel_stats<<<(int)(Lagg / 128), 256, 0, stream>>>(aggb, (int)Lagg, M1, s2, q2);
  fold_kernel<<<M2, 256, 0, stream>>>(Wg, bg, gamma2, beta2, s2, q2,
                                      (float)Lagg, M1, Wgf, bgf, 1);
  gemm_wmma_kernel<M1, 0, false><<<(int)(Lagg / 16 / 8), 256, 0, stream>>>(
      Wgf, aggb, nullptr, nullptr, bgf, nullptr, nullptr, outGf,
      M2, (int)Lagg, NPOINT);
}
